// HyenaFilter_22239340659187
// MI455X (gfx1250) — matrix operand order repaired and checked by reading, not compiled, not run on hardware
//
#include <hip/hip_runtime.h>
#include <hip/hip_bf16.h>
#include <math.h>
#include <stdint.h>

// Hyena filter + causal long conv for MI455X (gfx1250, wave32, WMMA).
//   B=4, C=768, L=8192, EMB=33, ORDER=64
//
// Kernel 1: implicit-filter MLP via V_WMMA_F32_16X16X32_F16 -> modulated
//           f16 kernel k[C][L] in workspace.
// Kernel 2: block-Toeplitz causal convolution via WMMA f16->f32.
//           One Toeplitz A fragment feeds 4 batches x 2 output tiles;
//           zero-padded LDS keeps every fragment build branch-free.
//           k row staged into LDS via the Tensor Data Mover (6-arg builtin).

#define BB   4
#define CC   768
#define LL   8192
#define EMBD 33
#define ORD  64

typedef __attribute__((ext_vector_type(16))) _Float16     v16h;
typedef __attribute__((ext_vector_type(8)))  float        v8f;
typedef __attribute__((ext_vector_type(4)))  unsigned int v4u;
typedef __attribute__((ext_vector_type(8)))  int          v8i;
typedef __attribute__((ext_vector_type(4)))  int          v4i;

#define KLEN (16 + LL)            // 16 zero pad: Toeplitz index >= -15
#define UROW (256 + LL)           // 256 zero pad: causal d > I blocks
#define CONV_LDS_BYTES ((KLEN + BB * UROW) * 2)   // 84000 B dynamic LDS

#if __has_builtin(__builtin_amdgcn_tensor_load_to_lds)
#define USE_TDM_K 1
#else
#define USE_TDM_K 0
#endif

// K index held by element e (0..15) of a 16-bit A fragment (16x32), for
// lane-half hi (contiguous order, matching the B fragments):
//   lanes 0-15 : K=0..15
//   lanes 16-31: K=16..31
__device__ __forceinline__ int a_kidx(int e, int hi) {
  return e + 16 * hi;
}
// B fragment (32x16, 16-bit): element e holds K = e + 16*hi, N = lane&15.

// ---------------------------------------------------------------------------
// Kernel 1: filter MLP.  One wave per 16 rows of L.
// ---------------------------------------------------------------------------
__global__ void __launch_bounds__(32)
hyena_filter_mlp(const float* __restrict__ z,
                 const float* __restrict__ deltas,
                 const float* __restrict__ W0, const float* __restrict__ b0,
                 const float* __restrict__ W1, const float* __restrict__ b1,
                 const float* __restrict__ W2, const float* __restrict__ b2,
                 const float* __restrict__ freq,
                 const float* __restrict__ Wout,
                 _Float16* __restrict__ kf16)
{
  __shared__ float    hbuf[16 * ORD];    // current activation tile [16][64]
  __shared__ float    zbuf[16 * ORD];    // z tile, zero-padded 33 -> 64
  __shared__ _Float16 w0buf[ORD * ORD];  // W0 zero-padded 33x64 -> 64x64
  const int lane = threadIdx.x;
  const int hi   = lane >> 4;
  const int lo   = lane & 15;
  const int l0   = blockIdx.x * 16;      // L-row base of this tile

  for (int idx = lane; idx < 16 * ORD; idx += 32) {
    const int row = idx >> 6, col = idx & 63;
    zbuf[idx] = (col < EMBD) ? z[(l0 + row) * EMBD + col] : 0.f;
  }
  for (int idx = lane; idx < ORD * ORD; idx += 32) {
    const int K = idx >> 6, nn2 = idx & 63;
    w0buf[idx] = (_Float16)((K < EMBD) ? W0[K * ORD + nn2] : 0.f);
  }
  __syncthreads();

  v16h a[2];
  #pragma unroll
  for (int kc = 0; kc < 2; ++kc)
    #pragma unroll
    for (int e = 0; e < 16; ++e)
      a[kc][e] = (_Float16)zbuf[lo * ORD + a_kidx(e, hi) + 32 * kc];

  for (int layer = 0; layer < 3; ++layer) {
    const float* W  = (layer == 1) ? W1 : W2;
    const float* bv = (layer == 0) ? b0 : ((layer == 1) ? b1 : b2);
    for (int nc = 0; nc < 4; ++nc) {
      const int nn = nc * 16 + lo;
      v8f acc = {};
      #pragma unroll
      for (int kc = 0; kc < 2; ++kc) {
        v16h bf;
        if (layer == 0) {
          #pragma unroll
          for (int e = 0; e < 16; ++e)
            bf[e] = w0buf[(e + 16 * hi + 32 * kc) * ORD + nn];
        } else {
          #pragma unroll
          for (int e = 0; e < 16; ++e)
            bf[e] = (_Float16)W[(e + 16 * hi + 32 * kc) * ORD + nn];
        }
        acc = __builtin_amdgcn_wmma_f32_16x16x32_f16(false, a[kc], false, bf,
                                                     (short)0, acc, false, false);
      }
      const float fr = freq[nn];
      const float bb = bv[nn];
      #pragma unroll
      for (int r = 0; r < 8; ++r)
        hbuf[(r + 8 * hi) * ORD + nn] = __sinf(fr * (acc[r] + bb));
    }
    __syncthreads();
    #pragma unroll
    for (int kc = 0; kc < 2; ++kc)
      #pragma unroll
      for (int e = 0; e < 16; ++e)
        a[kc][e] = (_Float16)hbuf[lo * ORD + a_kidx(e, hi) + 32 * kc];
    __syncthreads();
  }

  for (int cc = 0; cc < CC / 16; ++cc) {
    const int c = cc * 16 + lo;
    v8f acc = {};
    #pragma unroll
    for (int kc = 0; kc < 2; ++kc) {
      v16h bf;
      #pragma unroll
      for (int e = 0; e < 16; ++e)
        bf[e] = (_Float16)Wout[(e + 16 * hi + 32 * kc) * CC + c];
      acc = __builtin_amdgcn_wmma_f32_16x16x32_f16(false, a[kc], false, bf,
                                                   (short)0, acc, false, false);
    }
    const float dl = fabsf(deltas[c]);
    #pragma unroll
    for (int r = 0; r < 8; ++r) {
      const int   l  = l0 + r + 8 * hi;
      const float tt = (float)l * (1.f / (float)(LL - 1));
      const float v  = acc[r] * __expf(-tt * dl);
      kf16[(size_t)c * LL + l] = (_Float16)v;
    }
  }
}

// ---------------------------------------------------------------------------
// Kernel 2: causal block-Toeplitz convolution.
//   grid = (4 tile-groups, C), 128 threads (4 waves).  Wave w of group g
//   owns output tiles tA = 4g+w and tB = tA+16 for ALL 4 batches: each
//   Toeplitz A fragment feeds up to 8 WMMAs.
// ---------------------------------------------------------------------------
__global__ void __launch_bounds__(128)
hyena_conv(const float*    __restrict__ x,
           const _Float16* __restrict__ kf16,
           const float*    __restrict__ biasv,
           float*          __restrict__ out)
{
  extern __shared__ _Float16 smem[];     // dynamic: KLEN + BB*UROW halves
  _Float16* k_pad = smem;                // [KLEN]
  _Float16* u_pad = smem + KLEN;         // [BB][UROW]

  const int g   = blockIdx.x;            // 0..3
  const int c   = blockIdx.y;
  const int tid = threadIdx.x;
  const int n   = 1024 * g + 5120;       // staged prefix length (<= LL)

  const _Float16* kp = kf16 + (size_t)c * LL;

  // Zero pads.
  for (int i = tid; i < 256; i += 128) {
    if (i < 16) k_pad[i] = (_Float16)0.f;
    #pragma unroll
    for (int bb = 0; bb < BB; ++bb) u_pad[bb * UROW + i] = (_Float16)0.f;
  }

#if USE_TDM_K
  // Tensor Data Mover: DMA the f16 kernel row into LDS (1 x n tile, 2B elems).
  if (tid < 32) {                        // wave 0 issues; EXEC ignored by TDM
    const unsigned long long ga = (unsigned long long)(uintptr_t)kp;
    const unsigned lds_addr = (unsigned)(uintptr_t)(k_pad + 16);
    v4u g0 = { 1u,                                    // count=1 valid D#
               lds_addr,                              // lds_addr [63:32]
               (unsigned)(ga & 0xffffffffu),          // global_addr lo
               (unsigned)((ga >> 32) & 0x01ffffffu) | (2u << 30) }; // hi|type=2
    v8i g1 = { (int)0x00010000,                       // data_size=1 (2 bytes)
               (int)((n & 0xffff) << 16),             // tensor_dim0[15:0]
               (int)((unsigned)(n >> 16) | (1u << 16)), // dim0 hi | tensor_dim1=1
               (int)((n & 0xffff) << 16),             // tile_dim0 = n
               1,                                     // tile_dim1 = 1
               n,                                     // tensor_dim0_stride
               0, 0 };
    v4i gz4 = { 0, 0, 0, 0 };
    v8i gz8 = { 0, 0, 0, 0, 0, 0, 0, 0 };
    __builtin_amdgcn_tensor_load_to_lds(g0, g1, gz4, gz4, gz8, 0);
  }
#endif

  // Stage signal rows (f32 -> f16 conversion, so no DMA path here).
  for (int i = tid; i < n; i += 128) {
#if !USE_TDM_K
    k_pad[16 + i] = kp[i];
#endif
    #pragma unroll
    for (int bb = 0; bb < BB; ++bb) {
      const float* up = x + ((size_t)(bb * CC + c)) * LL;
      if (bb == 0) __builtin_prefetch(up + i + 1024, 0, 0);
      u_pad[bb * UROW + 256 + i] = (_Float16)up[i];
    }
  }
#if USE_TDM_K
  if (tid < 32) __builtin_amdgcn_s_wait_tensorcnt(0);
#endif
  __syncthreads();

  const int wid  = tid >> 5;
  const int lane = tid & 31;
  const int hi   = lane >> 4;
  const int lo   = lane & 15;
  const int tA   = 4 * g + wid;          // tile 0..15
  const int tB   = tA + 16;              // tile 16..31
  const int IA   = 16 * tA + lo;
  const int IB   = 16 * tB + lo;
  const int NK1  = 8 * (tA + 1);
  const int NK2  = 8 * (tB + 1);

  v8f accA[BB] = {};
  v8f accB[BB] = {};

  int kc = 0;
  // Phase 1: both tiles active.
  for (; kc < NK1; ++kc) {
    const int abase = 16 + kc * 32 + lo;
    v16h af;
    #pragma unroll
    for (int e = 0; e < 16; ++e) {
      const int ak = a_kidx(e, hi);
      af[e] = k_pad[abase + 16 * (ak >> 4) - (ak & 15)];
    }
    const int d    = 2 * kc + hi;
    const int offA = 256 + 16 * (IA - d);
    const int offB = 256 + 16 * (IB - d);
    #pragma unroll
    for (int bb = 0; bb < BB; ++bb) {
      const v16h bfA = *(const v16h*)(u_pad + bb * UROW + offA);
      accA[bb] = __builtin_amdgcn_wmma_f32_16x16x32_f16(false, af, false, bfA,
                                                        (short)0, accA[bb], false, false);
      const v16h bfB = *(const v16h*)(u_pad + bb * UROW + offB);
      accB[bb] = __builtin_amdgcn_wmma_f32_16x16x32_f16(false, af, false, bfB,
                                                        (short)0, accB[bb], false, false);
    }
  }
  // Phase 2: tile A complete (causality); tile B continues.
  for (; kc < NK2; ++kc) {
    const int abase = 16 + kc * 32 + lo;
    v16h af;
    #pragma unroll
    for (int e = 0; e < 16; ++e) {
      const int ak = a_kidx(e, hi);
      af[e] = k_pad[abase + 16 * (ak >> 4) - (ak & 15)];
    }
    const int offB = 256 + 16 * (IB - (2 * kc + hi));
    #pragma unroll
    for (int bb = 0; bb < BB; ++bb) {
      const v16h bfB = *(const v16h*)(u_pad + bb * UROW + offB);
      accB[bb] = __builtin_amdgcn_wmma_f32_16x16x32_f16(false, af, false, bfB,
                                                        (short)0, accB[bb], false, false);
    }
  }

  const float D = biasv[c];
  #pragma unroll
  for (int bb = 0; bb < BB; ++bb) {
    const float* up = x   + ((size_t)(bb * CC + c)) * LL;
    float*       op = out + ((size_t)(bb * CC + c)) * LL;
    #pragma unroll
    for (int r = 0; r < 8; ++r) {
      const int lA = 256 * tA + 16 * lo + r + 8 * hi;  // l = 16*(16t+N) + M
      const int lB = 256 * tB + 16 * lo + r + 8 * hi;
      op[lA] = accA[bb][r] + D * up[lA];
      op[lB] = accB[bb][r] + D * up[lB];
    }
  }
}

// ---------------------------------------------------------------------------
extern "C" void kernel_launch(void* const* d_in, const int* in_sizes, int n_in,
                              void* d_out, int out_size, void* d_ws, size_t ws_size,
                              hipStream_t stream) {
  const float* x      = (const float*)d_in[0];
  const float* biasv  = (const float*)d_in[1];
  const float* z      = (const float*)d_in[2];
  const float* deltas = (const float*)d_in[3];
  const float* W0     = (const float*)d_in[4];
  const float* b0     = (const float*)d_in[5];
  const float* W1     = (const float*)d_in[6];
  const float* b1     = (const float*)d_in[7];
  const float* W2     = (const float*)d_in[8];
  const float* b2     = (const float*)d_in[9];
  const float* freq   = (const float*)d_in[10];
  const float* Wout   = (const float*)d_in[11];

  _Float16* kf16 = (_Float16*)d_ws;            // C*L*2 = 12.6 MB scratch
  float*    out  = (float*)d_out;

  hyena_filter_mlp<<<dim3(LL / 16), dim3(32), 0, stream>>>(
      z, deltas, W0, b0, W1, b1, W2, b2, freq, Wout, kf16);

  hyena_conv<<<dim3(4, CC), dim3(128), CONV_LDS_BYTES, stream>>>(
      x, kf16, biasv, out);
}
